// DiscreteWaveletUpsample_76347338653735
// MI455X (gfx1250) — matrix operand order repaired and checked by reading, not compiled, not run on hardware
//
#include <hip/hip_runtime.h>

// ---------------------------------------------------------------------------
// DiscreteWaveletUpsample on MI455X (gfx1250), bf16 WMMA implicit-GEMM.
//   stage1: co = conv3x3(x, pre_w) + pre_b      (GEMM M=256, K=576, N=B*H*W)
//   stage2: Haar upsample folded into post-conv weights: per output parity
//           (P,Q) a dense GEMM  M=64, K=2*2*4*64=1024, N=B*H*W  over co.
// co is stored channel-last with a 1-pixel zero halo ring:
//   co_pad[b][hp=h+1 (130)][wp=w+1 (130)][ch (256)]  bf16
// so stage-2 B fragments are aligned 16B global loads (two per lane), no bounds.
// Main loops are barrier-free: A fragments come straight from L2-resident
// repacked weights; only stage-1 stages an x halo tile in LDS (one barrier).
// ---------------------------------------------------------------------------

typedef __bf16 v16bf __attribute__((ext_vector_type(16)));
typedef float  v8f   __attribute__((ext_vector_type(8)));

union Frag {
    v16bf v;
    uint4 q[2];
};

union Pack8 {
    unsigned short u[8];
    uint4 q;
};

__device__ __forceinline__ unsigned short f2bf(float f) {
    union { float f; unsigned u; } x; x.f = f;
    unsigned r = x.u + 0x7FFFu + ((x.u >> 16) & 1u);   // round-to-nearest-even
    return (unsigned short)(r >> 16);
}

// Problem sizes
#define BATCH 8
#define CIN   64
#define COUT  64
#define HH    128
#define WW    128
#define M1    256           // 4*COUT
#define K1    576           // 9*CIN
#define K2    1024          // 2*2*4*64
#define HP    130           // padded spatial
#define WP    130
#define CH    256           // co channels (channel-last stride)

// ---------------------------------------------------------------------------
// Kernel 0: weight prep.
//  Apre [256][576] bf16, k = (kh*3+kw)*64 + ci
//  Apost[4][64][1024] bf16, pq = P*2+Q, k = ((th*2+tw)*4 + s)*64 + ci
// ---------------------------------------------------------------------------
__global__ __launch_bounds__(256) void prep_kernel(
    const float* __restrict__ pre_w, const float* __restrict__ post_w,
    unsigned short* __restrict__ Apre, unsigned short* __restrict__ Apost)
{
    int idx = blockIdx.x * 256 + threadIdx.x;
    const int N1 = M1 * K1;
    if (idx < N1) {
        int m = idx / K1, k = idx % K1;
        int tap = k >> 6, ci = k & 63;
        int kh = tap / 3, kw = tap % 3;
        Apre[idx] = f2bf(pre_w[((m * CIN + ci) * 3 + kh) * 3 + kw]);
    } else {
        int d = idx - N1;
        if (d < 4 * COUT * K2) {
            int pq = d >> 16;             // 64*1024 per parity
            int m  = (d >> 10) & 63;
            int k  = d & 1023;
            int g  = k >> 6, ci = k & 63;
            int th = g >> 3, tw = (g >> 2) & 1, s = g & 3;
            int P  = pq >> 1, Q = pq & 1;
            int a  = s >> 1, bsel = s & 1; // haar row/col filter selectors
            float acc = 0.f;
            for (int dh = 0; dh < 3; ++dh) {
                int u    = P + dh - 1;
                int hoff = ((u + 2) >> 1) - 1;     // floor(u/2)
                if (hoff != th - 1 + P) continue;
                int p    = u & 1;
                float fh = 0.5f * ((a && p) ? -1.f : 1.f);
                for (int dw = 0; dw < 3; ++dw) {
                    int vq   = Q + dw - 1;
                    int woff = ((vq + 2) >> 1) - 1;
                    if (woff != tw - 1 + Q) continue;
                    int qq   = vq & 1;
                    float fw = 0.5f * ((bsel && qq) ? -1.f : 1.f);
                    acc += post_w[((m * COUT + ci) * 3 + dh) * 3 + dw] * fh * fw;
                }
            }
            Apost[d] = f2bf(acc);
        }
    }
}

// ---------------------------------------------------------------------------
// Kernel 0b: zero the 1-pixel halo ring of co_pad (runs every launch; ws is
// not re-poisoned between replays so this must be unconditional).
// Border pixels per batch: rows hp=0,129 (130 each) + cols wp=0,129 (128 each)
// = 516 pixels * 256 ch.
// ---------------------------------------------------------------------------
__global__ __launch_bounds__(256) void border_zero_kernel(unsigned short* __restrict__ co)
{
    int idx = blockIdx.x * 256 + threadIdx.x;
    const int perb = 516 * CH;
    if (idx >= BATCH * perb) return;
    int b   = idx / perb;
    int r   = idx % perb;
    int pix = r / CH;
    int ch  = r % CH;
    int hp, wp;
    if (pix < 130)      { hp = 0;             wp = pix; }
    else if (pix < 260) { hp = 129;           wp = pix - 130; }
    else if (pix < 388) { hp = pix - 260 + 1; wp = 0; }
    else                { hp = pix - 388 + 1; wp = 129; }
    co[(((size_t)b * HP + hp) * WP + wp) * CH + ch] = 0;
}

// ---------------------------------------------------------------------------
// Kernel 1: pre-conv implicit GEMM.  Block = 32-pixel row segment of one (b,h).
//  8 waves, each owns M=32 x N=32 (2x2 wmma tiles).  K loop 18 x 32.
//  A fragments straight from global (L2-resident); B from LDS x halo tile.
// ---------------------------------------------------------------------------
__global__ __launch_bounds__(256) void preconv_kernel(
    const float* __restrict__ x, const unsigned short* __restrict__ Apre,
    const float* __restrict__ pre_b, unsigned short* __restrict__ co)
{
    __shared__ unsigned short sX[3 * 34 * 64];        // halo tile, ch-contig, 13056 B

    int blk = blockIdx.x;                 // 0..4095 = b*512 + h*4 + wseg
    int b   = blk >> 9;
    int rem = blk & 511;
    int h   = rem >> 2;
    int w0  = (rem & 3) << 5;
    int tid = threadIdx.x;

    // stage x halo tile: sX[(hr*34 + wl)*64 + ci] = bf16(x[b,ci,h+hr-1,w0+wl-1])
    for (int e = tid; e < 3 * 34 * 64; e += 256) {
        int wl = e % 34;
        int t2 = e / 34;
        int hr = t2 % 3;
        int ci = t2 / 3;
        int hh = h + hr - 1;
        int wg = w0 + wl - 1;
        float v = 0.f;
        if (hh >= 0 && hh < HH && wg >= 0 && wg < WW)
            v = x[(((b * CIN) + ci) * HH + hh) * WW + wg];
        sX[(hr * 34 + wl) * 64 + ci] = f2bf(v);
    }
    __syncthreads();

    int wv = tid >> 5, lane = tid & 31;
    int lm = lane & 15, hilo = lane >> 4;
    int mt0 = wv * 32;

    v8f acc[4];
    for (int i = 0; i < 4; ++i)
        for (int j = 0; j < 8; ++j) acc[i][j] = 0.f;

    for (int kt = 0; kt < 18; ++kt) {
        int k0  = kt * 32;
        int tap = kt >> 1, ci0 = (kt & 1) << 5;
        int kh = tap / 3, kw = tap % 3;

        Frag afr[2], bfr[2];
        for (int i = 0; i < 2; ++i) {           // A: lane=M, K halves 8 | 8
            const unsigned short* ap = Apre + (mt0 + i * 16 + lm) * K1 + k0 + hilo * 8;
            afr[i].q[0] = *(const uint4*)(ap);
            afr[i].q[1] = *(const uint4*)(ap + 16);
        }
        for (int j = 0; j < 2; ++j) {           // B: lane=N, K halves 8 | 8 (=ci)
            int n = j * 16 + lm;
            const unsigned short* bp = sX + ((kh * 34) + n + kw) * 64 + ci0 + hilo * 8;
            bfr[j].q[0] = *(const uint4*)(bp);
            bfr[j].q[1] = *(const uint4*)(bp + 16);
        }
        for (int i = 0; i < 2; ++i)
            for (int j = 0; j < 2; ++j)
                acc[i * 2 + j] = __builtin_amdgcn_wmma_f32_16x16x32_bf16(
                    false, afr[i].v, false, bfr[j].v,
                    (short)0, acc[i * 2 + j], false, false);
    }

    // epilogue: bias + bf16, channel-last contiguous 16B stores into co_pad
    int hp = h + 1;
    for (int i = 0; i < 2; ++i)
        for (int j = 0; j < 2; ++j) {
            int n = j * 16 + lm;
            size_t pixbase = (((size_t)b * HP + hp) * WP + (w0 + n + 1)) * CH;
            int mbase = mt0 + i * 16 + hilo * 8;
            Pack8 pk;
            for (int v = 0; v < 8; ++v)
                pk.u[v] = f2bf(acc[i * 2 + j][v] + pre_b[mbase + v]);
            *(uint4*)(co + pixbase + mbase) = pk.q;
        }
}

// ---------------------------------------------------------------------------
// Kernel 2: Haar-folded post-conv GEMM.  Block = (b, h0, P); 8 waves =
//  4 M-tiles x 2 Q-parities, each wave N=128 (8 wmma tiles).  K loop 32 x 32
//  over k = ((th*2+tw)*4+s)*64+ci.  Zero LDS, zero barriers: A and B
//  fragments are aligned direct global loads (co halo ring removes bounds).
// ---------------------------------------------------------------------------
__global__ __launch_bounds__(256) void postconv_kernel(
    const unsigned short* __restrict__ co, const unsigned short* __restrict__ Apost,
    const float* __restrict__ post_b, float* __restrict__ out)
{
    int blk = blockIdx.x;                  // 0..2047 = b*256 + h0*2 + P
    int b   = blk >> 8;
    int rem = blk & 255;
    int h0  = rem >> 1;
    int P   = rem & 1;
    int tid = threadIdx.x;

    int wv = tid >> 5, lane = tid & 31;
    int lm = lane & 15, hilo = lane >> 4;
    int mt = (wv & 3) * 16;
    int Q  = wv >> 2;
    int pq = P * 2 + Q;

    v8f acc[8];
    for (int i = 0; i < 8; ++i)
        for (int j = 0; j < 8; ++j) acc[i][j] = 0.f;

    const unsigned short* arow = Apost + (pq * COUT + mt + lm) * K2;

    for (int kt = 0; kt < 32; ++kt) {
        int k0  = kt * 32;
        int g   = kt >> 1;
        int th  = g >> 3, tw = (g >> 2) & 1, s = g & 3;
        int ci0 = (kt & 1) << 5;

        Frag afr;
        {
            const unsigned short* ap = arow + k0 + hilo * 8;
            afr.q[0] = *(const uint4*)(ap);
            afr.q[1] = *(const uint4*)(ap + 16);
        }
        int hp   = h0 + th + P;            // always in [0,129]
        int chof = s * 64 + ci0 + hilo * 8;
        const unsigned short* brow = co + (((size_t)b * HP + hp) * WP) * CH + chof;
        for (int j = 0; j < 8; ++j) {
            int wp = j * 16 + lm + tw + Q; // always in [0,129]
            Frag bfr;
            const unsigned short* bp = brow + (size_t)wp * CH;
            bfr.q[0] = *(const uint4*)(bp);
            bfr.q[1] = *(const uint4*)(bp + 16);
            acc[j] = __builtin_amdgcn_wmma_f32_16x16x32_bf16(
                false, afr.v, false, bfr.v, (short)0, acc[j], false, false);
        }
    }

    // epilogue: out[b][m][2*h0+P][2*w0+Q]
    int H2 = 2 * h0 + P;
    for (int j = 0; j < 8; ++j) {
        int w0p = j * 16 + lm;
        int W2  = 2 * w0p + Q;
        for (int v = 0; v < 8; ++v) {
            int m = mt + v + hilo * 8;
            out[(((b * COUT) + m) * 256 + H2) * 256 + W2] = acc[j][v] + post_b[m];
        }
    }
}

// ---------------------------------------------------------------------------
extern "C" void kernel_launch(void* const* d_in, const int* in_sizes, int n_in,
                              void* d_out, int out_size, void* d_ws, size_t ws_size,
                              hipStream_t stream)
{
    (void)in_sizes; (void)n_in; (void)out_size; (void)ws_size;
    const float* x      = (const float*)d_in[0];
    const float* pre_w  = (const float*)d_in[1];
    const float* pre_b  = (const float*)d_in[2];
    const float* post_w = (const float*)d_in[3];
    const float* post_b = (const float*)d_in[4];
    float* out = (float*)d_out;

    char* ws = (char*)d_ws;
    unsigned short* Apre  = (unsigned short*)ws;                      // 294912 B
    unsigned short* Apost = (unsigned short*)(ws + 294912);           // 524288 B
    unsigned short* co    = (unsigned short*)(ws + 294912 + 524288);  // 69.2 MB bf16 padded

    const int prep_elems = M1 * K1 + 4 * COUT * K2;   // 147456 + 262144
    prep_kernel<<<dim3((prep_elems + 255) / 256), dim3(256), 0, stream>>>(
        pre_w, post_w, Apre, Apost);

    const int border_elems = BATCH * 516 * CH;        // 1,056,768
    border_zero_kernel<<<dim3((border_elems + 255) / 256), dim3(256), 0, stream>>>(co);

    preconv_kernel<<<dim3(BATCH * HH * 4), dim3(256), 0, stream>>>(
        x, Apre, pre_b, co);
    postconv_kernel<<<dim3(BATCH * HH * 2), dim3(256), 0, stream>>>(
        co, Apost, post_b, out);
}
